// HeteroVisitGNN_62122406969960
// MI455X (gfx1250) — matrix operand order repaired and checked by reading, not compiled, not run on hardware
//
#include <hip/hip_runtime.h>
#include <math.h>

// ---------------------------------------------------------------------------
// HeteroVisitGNN on gfx1250: WMMA bf16 GEMMs (A-reuse across col tiles,
// fully unrolled K, B fragments pre-swizzled in LDS for b128 loads)
// + L2-resident edge scatter with float atomics.
// ---------------------------------------------------------------------------

#define NVN 100000
#define NEN 50000
#define EDG 600000
#define HDIM 64

typedef __attribute__((ext_vector_type(16))) __bf16 v16bf;
typedef __attribute__((ext_vector_type(8)))  float  v8f;

static __device__ __forceinline__ void atomicMaxF(float* addr, float v) {
    // monotonic-map trick; buffer is initialized to -inf
    if (v >= 0.f) atomicMax((int*)addr, __float_as_int(v));
    else          atomicMin((unsigned int*)addr, __float_as_uint(v));
}

// ---------------------------------------------------------------------------
// out[N,Ho] = X[N,K] @ W[K,Ho] (+ bias).
// One wave per 16-row strip; the A fragment is reused across all Ho/16
// column tiles. W is staged into LDS pre-swizzled into the WMMA B-fragment
// lane layout, so each fragment is one 32-byte vector LDS load per lane.
// ---------------------------------------------------------------------------
template <int K, int Ho>
__global__ void gemm_bias_wmma(const float* __restrict__ X,
                               const float* __restrict__ W,
                               const float* __restrict__ bias,
                               float* __restrict__ out, int N)
{
    constexpr int CT = Ho / 16;            // column tiles per wave
    constexpr int KB = K / 32;             // k-steps
    // swizzled layout: [(kb*CT + ct)*512 + lane*16 + i]
    //   element i of lane L = W[kb*32 + (i&7) + 8*(L>>4) + 16*(i>>3), ct*16 + (L&15)]
    __shared__ __bf16 sW[K * Ho];

    for (int idx = threadIdx.x; idx < K * Ho; idx += blockDim.x) {
        const int kb  = idx / (CT * 512);
        const int rem = idx - kb * (CT * 512);
        const int ct  = rem >> 9;
        const int r2  = rem & 511;
        const int ln  = r2 >> 4;
        const int i   = r2 & 15;
        const int srcK = kb * 32 + (i & 7) + 8 * (ln >> 4) + 16 * (i >> 3);
        const int srcN = ct * 16 + (ln & 15);
        sW[idx] = (__bf16)W[srcK * Ho + srcN];     // hw cvt f32->bf16
    }
    __syncthreads();

    const int lane = threadIdx.x & 31;
    const int half = lane >> 4;            // 0: K-lo half, 1: K-hi half
    const int l16  = lane & 15;

    const int rowTile = blockIdx.x * (blockDim.x >> 5) + (threadIdx.x >> 5);
    if (rowTile * 16 >= N) return;         // wave-uniform, after barrier
    const int row0 = rowTile * 16;

    v8f acc[CT];
#pragma unroll
    for (int ct = 0; ct < CT; ++ct) acc[ct] = (v8f){};

#pragma unroll
    for (int kb = 0; kb < KB; ++kb) {
        const int k0 = kb * 32;
        // ---- A fragment (global, b128 loads): row = row0+l16,
        //      K(i) = k0 + (i&7) + 16*(i>>3) + 8*half
        const float4* Arow =
            reinterpret_cast<const float4*>(X + (size_t)(row0 + l16) * K + k0 + 8 * half);
        const float4 p0 = Arow[0];
        const float4 p1 = Arow[1];
        const float4 p2 = Arow[4];         // +16 floats
        const float4 p3 = Arow[5];
        float av[16] = {p0.x, p0.y, p0.z, p0.w, p1.x, p1.y, p1.z, p1.w,
                        p2.x, p2.y, p2.z, p2.w, p3.x, p3.y, p3.z, p3.w};
        v16bf fa;
#pragma unroll
        for (int i = 0; i < 16; ++i) fa[i] = (__bf16)av[i];

        // ---- one WMMA per column tile, same A; fb = single 32B LDS load
#pragma unroll
        for (int ct = 0; ct < CT; ++ct) {
            const v16bf fb = *reinterpret_cast<const v16bf*>(
                sW + ((kb * CT + ct) << 9) + (lane << 4));
            acc[ct] = __builtin_amdgcn_wmma_f32_16x16x32_bf16(
                          false, fa, false, fb, (short)0, acc[ct], false, false);
        }
    }

#pragma unroll
    for (int ct = 0; ct < CT; ++ct) {
        const float bv = bias ? bias[ct * 16 + l16] : 0.f;
#pragma unroll
        for (int r = 0; r < 8; ++r) {
            const int m = row0 + r + 8 * half;
            out[(size_t)m * Ho + ct * 16 + l16] = acc[ct][r] + bv;
        }
    }
}

// alpha[n] = dot(X[n, 0:64], v[0:64])  (one wave32 per node, shuffle reduce)
__global__ void rowdot64(const float* __restrict__ X, const float* __restrict__ v,
                         float* __restrict__ out, int N)
{
    const int wave = (int)((blockIdx.x * blockDim.x + threadIdx.x) >> 5);
    const int lane = threadIdx.x & 31;
    if (wave >= N) return;
    float t = X[(size_t)wave * HDIM + lane]      * v[lane]
            + X[(size_t)wave * HDIM + lane + 32] * v[lane + 32];
#pragma unroll
    for (int o = 16; o; o >>= 1) t += __shfl_xor(t, o, 32);
    if (lane == 0) out[wave] = t;
}

// wad[k] = sum_j W[k,j] * att[j]   (tiny: 64 threads)
__global__ void watt_kernel(const float* __restrict__ W,
                            const float* __restrict__ att,
                            float* __restrict__ wad)
{
    const int k = threadIdx.x;
    float s = 0.f;
    for (int j = 0; j < HDIM; ++j) s += W[k * HDIM + j] * att[j];
    wad[k] = s;
}

__global__ void fill_f32(float* __restrict__ p, float v, int n)
{
    const int i = blockIdx.x * blockDim.x + threadIdx.x;
    if (i < n) p[i] = v;
}

// out[n*64+f] = b1[f] (+ b2[f])   -- HeteroConv sum of GAT biases
__global__ void init_bias(float* __restrict__ out, const float* __restrict__ b1,
                          const float* __restrict__ b2, int total)
{
    const int i = blockIdx.x * blockDim.x + threadIdx.x;
    if (i >= total) return;
    float v = b1[i & 63];
    if (b2) v += b2[i & 63];
    out[i] = v;
}

__global__ void relu_inplace(float* __restrict__ p, int n)
{
    const int i = blockIdx.x * blockDim.x + threadIdx.x;
    if (i < n) p[i] = fmaxf(p[i], 0.f);
}

// pass 1: a = leaky_relu(asrc[src] + adst[dst]); segment max into m
__global__ void edge_a_max(const int* __restrict__ src, const int* __restrict__ dst,
                           const float* __restrict__ asrc, const float* __restrict__ adst,
                           float* __restrict__ aedge, float* __restrict__ m, int E)
{
    const int e = blockIdx.x * blockDim.x + threadIdx.x;
    if (e >= E) return;
    float a = asrc[src[e]];
    if (adst) a += adst[dst[e]];
    a = (a > 0.f) ? a : 0.2f * a;
    aedge[e] = a;
    atomicMaxF(&m[dst[e]], a);
}

// pass 2: e = exp(a - m[dst]); segment sum into s
__global__ void edge_exp_sum(const int* __restrict__ dst, float* __restrict__ aedge,
                             const float* __restrict__ m, float* __restrict__ s, int E)
{
    const int e = blockIdx.x * blockDim.x + threadIdx.x;
    if (e >= E) return;
    const int d = dst[e];
    const float v = __expf(aedge[e] - m[d]);
    aedge[e] = v;
    atomicAdd(&s[d], v);
}

// pass 2.5: coef = e / s[dst]  (so the 64-wide scatter doesn't re-gather s)
__global__ void edge_coef(const int* __restrict__ dst, float* __restrict__ eexp,
                          const float* __restrict__ s, int E)
{
    const int e = blockIdx.x * blockDim.x + threadIdx.x;
    if (e >= E) return;
    eexp[e] = eexp[e] / s[dst[e]];
}

// pass 3: out[dst] += h[src] * coef[e]   (64 threads per edge)
__global__ void edge_aggregate(const int* __restrict__ src, const int* __restrict__ dst,
                               const float* __restrict__ h, const float* __restrict__ coef,
                               float* __restrict__ out, int E)
{
    const long long idx = (long long)blockIdx.x * blockDim.x + threadIdx.x;
    const int e = (int)(idx >> 6);
    const int f = (int)(idx & 63);
    if (e >= E) return;
    const float c = coef[e];
    atomicAdd(&out[(size_t)dst[e] * HDIM + f], h[(size_t)src[e] * HDIM + f] * c);
}

// ---------------------------------------------------------------------------
// Host orchestration
// ---------------------------------------------------------------------------
template <int K, int Ho>
static inline void launch_gemm(const float* X, const float* W, const float* b,
                               float* out, int N, hipStream_t st)
{
    const int waves  = N / 16;             // one wave per 16-row strip
    const int blocks = (waves + 7) / 8;
    gemm_bias_wmma<K, Ho><<<blocks, 256, 0, st>>>(X, W, b, out, N);
}

static void run_gat(const float* x_src, int Ns,
                    const float* x_dst, int Nd,       // x_dst may be null
                    const float* W, const float* att_s, const float* att_d,
                    const int* src, const int* dst, int E,
                    float* h, float* asrc, float* adst, float* wad,
                    float* m, float* s, float* eexp,
                    float* out_accum, hipStream_t st)
{
    launch_gemm<HDIM, HDIM>(x_src, W, nullptr, h, Ns, st);
    rowdot64<<<(Ns + 7) / 8, 256, 0, st>>>(h, att_s, asrc, Ns);
    if (x_dst) {
        watt_kernel<<<1, 64, 0, st>>>(W, att_d, wad);
        rowdot64<<<(Nd + 7) / 8, 256, 0, st>>>(x_dst, wad, adst, Nd);
    }
    fill_f32<<<(Nd + 255) / 256, 256, 0, st>>>(m, -INFINITY, Nd);
    fill_f32<<<(Nd + 255) / 256, 256, 0, st>>>(s, 0.f, Nd);
    edge_a_max<<<(E + 255) / 256, 256, 0, st>>>(src, dst, asrc,
                                                x_dst ? adst : nullptr, eexp, m, E);
    edge_exp_sum<<<(E + 255) / 256, 256, 0, st>>>(dst, eexp, m, s, E);
    edge_coef<<<(E + 255) / 256, 256, 0, st>>>(dst, eexp, s, E);
    edge_aggregate<<<(unsigned)(((long long)E * 64 + 255) / 256), 256, 0, st>>>(
        src, dst, h, eexp, out_accum, E);
}

extern "C" void kernel_launch(void* const* d_in, const int* in_sizes, int n_in,
                              void* d_out, int out_size, void* d_ws, size_t ws_size,
                              hipStream_t stream)
{
    (void)in_sizes; (void)n_in; (void)out_size; (void)ws_size;

    // --- input unpacking (setup_inputs dict order, params flattened in
    //     insertion order: lin_v{W,b}, lin_e{W,b},
    //     conv1{vv,vvr,ve,ev}x{W,att_src,att_dst,bias}, conv2 same, lin_out{W,b})
    const float* x_visit  = (const float*)d_in[0];
    const float* x_entity = (const float*)d_in[1];
    const float* linv_W = (const float*)d_in[2];
    const float* linv_b = (const float*)d_in[3];
    const float* line_W = (const float*)d_in[4];
    const float* line_b = (const float*)d_in[5];
    const float* const* P = (const float* const*)d_in;
    #define GAT_W(base)  P[(base) + 0]
    #define GAT_AS(base) P[(base) + 1]
    #define GAT_AD(base) P[(base) + 2]
    #define GAT_B(base)  P[(base) + 3]
    const int C1VV = 6, C1VVR = 10, C1VE = 14, C1EV = 18;
    const int C2VV = 22, C2EV = 34;                  // layer-2 ve/vvr are dead code
    const float* lout_W = (const float*)d_in[38];
    const float* lout_b = (const float*)d_in[39];
    const int* vv_src  = (const int*)d_in[40];
    const int* vv_dst  = (const int*)d_in[41];
    const int* vvr_src = (const int*)d_in[42];
    const int* vvr_dst = (const int*)d_in[43];
    const int* ve_src  = (const int*)d_in[44];
    const int* ve_dst  = (const int*)d_in[45];
    const int* ev_src  = (const int*)d_in[46];
    const int* ev_dst  = (const int*)d_in[47];

    // --- workspace carve-up (floats)
    float* ws = (float*)d_ws;
    size_t o = 0;
    auto alloc = [&](size_t n) { float* p = ws + o; o += n; return p; };
    float* xvA  = alloc((size_t)NVN * HDIM);
    float* xeA  = alloc((size_t)NEN * HDIM);
    float* xvB  = alloc((size_t)NVN * HDIM);
    float* xeB  = alloc((size_t)NEN * HDIM);
    float* xvrB = alloc((size_t)NVN * HDIM);
    float* h    = alloc((size_t)NVN * HDIM);
    float* asrc = alloc(NVN);
    float* adst = alloc(NVN);
    float* wad  = alloc(HDIM);
    float* m    = alloc(NVN);
    float* s    = alloc(NVN);
    float* eexp = alloc(EDG);

    // --- input projections
    launch_gemm<256, HDIM>(x_visit,  linv_W, linv_b, xvA, NVN, stream);
    launch_gemm<128, HDIM>(x_entity, line_W, line_b, xeA, NEN, stream);

    // --- layer 1 (accumulate into B set; xvr dst features absent -> null)
    init_bias<<<(NVN * HDIM + 255) / 256, 256, 0, stream>>>(xvB,  GAT_B(C1VV), GAT_B(C1EV), NVN * HDIM);
    init_bias<<<(NEN * HDIM + 255) / 256, 256, 0, stream>>>(xeB,  GAT_B(C1VE), nullptr,     NEN * HDIM);
    init_bias<<<(NVN * HDIM + 255) / 256, 256, 0, stream>>>(xvrB, GAT_B(C1VVR), nullptr,    NVN * HDIM);

    run_gat(xvA, NVN, xvA, NVN, GAT_W(C1VV),  GAT_AS(C1VV),  GAT_AD(C1VV),
            vv_src,  vv_dst,  EDG, h, asrc, adst, wad, m, s, eexp, xvB,  stream);
    run_gat(xeA, NEN, xvA, NVN, GAT_W(C1EV),  GAT_AS(C1EV),  GAT_AD(C1EV),
            ev_src,  ev_dst,  EDG, h, asrc, adst, wad, m, s, eexp, xvB,  stream);
    run_gat(xvA, NVN, xeA, NEN, GAT_W(C1VE),  GAT_AS(C1VE),  GAT_AD(C1VE),
            ve_src,  ve_dst,  EDG, h, asrc, adst, wad, m, s, eexp, xeB,  stream);
    run_gat(xvA, NVN, nullptr, NVN, GAT_W(C1VVR), GAT_AS(C1VVR), GAT_AD(C1VVR),
            vvr_src, vvr_dst, EDG, h, asrc, adst, wad, m, s, eexp, xvrB, stream);

    relu_inplace<<<(NVN * HDIM + 255) / 256, 256, 0, stream>>>(xvB,  NVN * HDIM);
    relu_inplace<<<(NEN * HDIM + 255) / 256, 256, 0, stream>>>(xeB,  NEN * HDIM);
    relu_inplace<<<(NVN * HDIM + 255) / 256, 256, 0, stream>>>(xvrB, NVN * HDIM);

    // --- layer 2: only the visit-dst output feeds lin_out -> compute vv + ev only
    init_bias<<<(NVN * HDIM + 255) / 256, 256, 0, stream>>>(xvA, GAT_B(C2VV), GAT_B(C2EV), NVN * HDIM);
    run_gat(xvB, NVN, xvB, NVN, GAT_W(C2VV), GAT_AS(C2VV), GAT_AD(C2VV),
            vv_src, vv_dst, EDG, h, asrc, adst, wad, m, s, eexp, xvA, stream);
    run_gat(xeB, NEN, xvB, NVN, GAT_W(C2EV), GAT_AS(C2EV), GAT_AD(C2EV),
            ev_src, ev_dst, EDG, h, asrc, adst, wad, m, s, eexp, xvA, stream);
    relu_inplace<<<(NVN * HDIM + 255) / 256, 256, 0, stream>>>(xvA, NVN * HDIM);

    // --- output projection: [NV,64] @ [64,128] + b
    launch_gemm<HDIM, 128>(xvA, lout_W, lout_b, (float*)d_out, NVN, stream);
}
